// TokenCodebook_40389872452007
// MI455X (gfx1250) — matrix operand order repaired and checked by reading, not compiled, not run on hardware
//
#include <hip/hip_runtime.h>
#include <hip/hip_bf16.h>

typedef __attribute__((ext_vector_type(16))) _Float16 v16h;
typedef __attribute__((ext_vector_type(8)))  _Float16 v8h;
typedef __attribute__((ext_vector_type(8)))  float    v8f;
typedef __attribute__((ext_vector_type(4)))  unsigned int u32x4;
typedef __attribute__((ext_vector_type(8)))  unsigned int u32x8;

#define EDIM   256
#define MTILE  64
#define WAVES  4
#define NSPLIT 16
#define TOPK   8

// ---------------------------------------------------------------------------
// Kernel 1: L2-normalize f32 rows of length 256 -> f16 rows. One wave per row.
// Rows in [rows, rows_pad) are zero-filled padding.
// ---------------------------------------------------------------------------
__global__ __launch_bounds__(256) void normalize_f16_kernel(
    const float* __restrict__ src, _Float16* __restrict__ dst,
    int rows, int rows_pad)
{
    const int gw   = (blockIdx.x * blockDim.x + threadIdx.x) >> 5; // global wave = row
    const int lane = threadIdx.x & 31;
    if (gw >= rows_pad) return;

    v8h out;
    if (gw < rows) {
        const float* rp = src + (size_t)gw * EDIM + lane * 8;
        float4 a = *(const float4*)(rp);
        float4 b = *(const float4*)(rp + 4);
        float ss = a.x*a.x + a.y*a.y + a.z*a.z + a.w*a.w
                 + b.x*b.x + b.y*b.y + b.z*b.z + b.w*b.w;
        #pragma unroll
        for (int s = 16; s > 0; s >>= 1) ss += __shfl_xor(ss, s, 32);
        const float inv = 1.0f / fmaxf(sqrtf(ss), 1e-12f);
        out[0] = (_Float16)(a.x*inv); out[1] = (_Float16)(a.y*inv);
        out[2] = (_Float16)(a.z*inv); out[3] = (_Float16)(a.w*inv);
        out[4] = (_Float16)(b.x*inv); out[5] = (_Float16)(b.y*inv);
        out[6] = (_Float16)(b.z*inv); out[7] = (_Float16)(b.w*inv);
    } else {
        #pragma unroll
        for (int j = 0; j < 8; ++j) out[j] = (_Float16)0.0f;
    }
    *(v8h*)(dst + (size_t)gw * EDIM + lane * 8) = out;
}

// ---------------------------------------------------------------------------
// TDM: async DMA of a 16-row x 256-half 2D tile of the f16 emb matrix into
// LDS. D# built per CDNA5 ISA §8 (Group0 128b + Group1 256b, 2D form).
// Issued by one wave; completion via s_wait_tensorcnt + workgroup barrier.
// ---------------------------------------------------------------------------
__device__ __forceinline__ void tdm_load_b_tile(const _Float16* gbase, int vbase,
                                                unsigned lds_off, int VP)
{
    const unsigned long long ga =
        (unsigned long long)(size_t)gbase + (unsigned long long)vbase * (EDIM * 2);
    u32x4 g0;
    g0[0] = 1u;                                   // count=1 user descriptor
    g0[1] = lds_off;                              // lds_addr (bytes)
    g0[2] = (unsigned)(ga & 0xFFFFFFFFu);         // global_addr[31:0]
    g0[3] = (unsigned)((ga >> 32) & 0x01FFFFFFu)  // global_addr[56:32]
          | (2u << 30);                           // type=2 ("image")
    u32x8 g1;
    g1[0] = 1u << 16;                             // wg_mask=0, data_size=1 (2B)
    g1[1] = ((unsigned)EDIM & 0xFFFFu) << 16;     // tensor_dim0[15:0]=256
    g1[2] = ((unsigned)VP   & 0xFFFFu) << 16;     // tensor_dim0 hi=0 | tensor_dim1 lo
    g1[3] = ((unsigned)VP >> 16)                  // tensor_dim1 hi
          | (((unsigned)EDIM & 0xFFFFu) << 16);   // tile_dim0=256
    g1[4] = 16u;                                  // tile_dim1=16 rows, tile_dim2=0
    g1[5] = (unsigned)EDIM;                       // tensor_dim0_stride=256
    g1[6] = 0u;                                   // stride hi / dim1_stride lo
    g1[7] = 0u;
    asm volatile("tensor_load_to_lds %0, %1" :: "s"(g0), "s"(g1) : "memory");
}

// ---------------------------------------------------------------------------
// Kernel 2: fused WMMA GEMM (sim = A * B^T) + running top-8 per row.
// Workgroup = 4 waves = 64-row M-tile; blockIdx.y splits V into 16 ranges.
// Each 16-column emb tile is TDM-DMAed into a double-buffered LDS slot and
// shared by all 4 waves; the DMA overlaps the previous tile's WMMA chain.
// ---------------------------------------------------------------------------
__global__ __launch_bounds__(128) void gemm_topk_kernel(
    const _Float16* __restrict__ A, const _Float16* __restrict__ Bm,
    float* __restrict__ pvals, int* __restrict__ pidx,
    int V, int VP, int ntiles)
{
    __shared__ _Float16 ldsA[MTILE * EDIM];        // 32 KB
    __shared__ _Float16 ldsB[2][16 * EDIM];        // 16 KB (double-buffered)
    __shared__ float    sims[WAVES][16][16];       //  4 KB

    const int tid   = threadIdx.x;
    const int lane  = tid & 31;
    const int wave  = tid >> 5;
    const int mbase = blockIdx.x * MTILE;
    const int split = blockIdx.y;

    const unsigned ldsB_off[2] = { (unsigned)(size_t)&ldsB[0][0],
                                   (unsigned)(size_t)&ldsB[1][0] };

    // ---- stage A tile (64 x 256 halves = 32 KB), 16-byte chunks ----
    {
        const uint4* gA = (const uint4*)(A + (size_t)mbase * EDIM);
        uint4* sA = (uint4*)ldsA;
        #pragma unroll
        for (int c = 0; c < (MTILE * EDIM * 2 / 16) / 128; ++c)
            sA[tid + c * 128] = gA[tid + c * 128];
    }

    // ---- per-lane running top-8 (lanes 0..15 own one row each) ----
    float tv[TOPK]; int ti[TOPK];
    #pragma unroll
    for (int j = 0; j < TOPK; ++j) { tv[j] = -2.0f; ti[j] = 0; }

    const int tps    = (ntiles + NSPLIT - 1) / NSPLIT;
    const int tstart = split * tps;
    const int tend   = (tstart + tps < ntiles) ? (tstart + tps) : ntiles;

    // fragment addressing (per ISA VGPR layouts)
    const int mrow   = wave * 16 + (lane & 15);
    const int koffA  = (lane >> 4) << 3;   // A: 0 or 8  (+16 for upper half)
    const int nrow   = lane & 15;
    const int koffB  = (lane >> 4) << 3;   // B: 0 or 8  (+16 for upper half)

    // prologue: kick off DMA of the first B tile
    if (wave == 0 && tstart < tend)
        tdm_load_b_tile(Bm, tstart * 16, ldsB_off[0], VP);

    for (int t = tstart; t < tend; ++t) {
        const int vbase = t * 16;
        const int buf   = (t - tstart) & 1;

        if (wave == 0)
            asm volatile("s_wait_tensorcnt 0x0" ::: "memory"); // tile t landed
        __syncthreads();  // publish tile t; all waves done reading buf^1

        if (wave == 0 && t + 1 < tend)                         // DMA tile t+1
            tdm_load_b_tile(Bm, (t + 1) * 16, ldsB_off[buf ^ 1], VP);
        if (t + 2 < tend) {                                    // warm L2 ahead
            const char* nb = (const char*)(Bm + (size_t)(vbase + 32) * EDIM);
            __builtin_prefetch(nb + tid * 64, 0, 1);
        }

        // ---- 16x16x256 accumulation: 8 x v_wmma_f32_16x16x32_f16 ----
        v8f acc = {};
        #pragma unroll
        for (int k0 = 0; k0 < EDIM; k0 += 32) {
            const _Float16* ap = &ldsA[mrow * EDIM + k0 + koffA];
            const _Float16* bp = &ldsB[buf][nrow * EDIM + k0 + koffB];
            v8h alo = *(const v8h*)(ap);
            v8h ahi = *(const v8h*)(ap + 16);
            v8h blo = *(const v8h*)(bp);
            v8h bhi = *(const v8h*)(bp + 16);
            v16h af, bf;
            #pragma unroll
            for (int j = 0; j < 8; ++j) {
                af[j] = alo[j]; af[8 + j] = ahi[j];
                bf[j] = blo[j]; bf[8 + j] = bhi[j];
            }
            acc = __builtin_amdgcn_wmma_f32_16x16x32_f16(
                false, af, false, bf, (short)0, acc, false, false);
        }

        // ---- scatter C fragment to LDS: VGPR r -> rows r / r+8 ----
        {
            const int n   = lane & 15;
            const int mhi = (lane >> 4) * 8;
            #pragma unroll
            for (int r = 0; r < 8; ++r)
                sims[wave][mhi + r][n] = acc[r];
        }
        asm volatile("s_wait_dscnt 0" ::: "memory");

        // ---- lanes 0..15: fold 16 candidates into sorted top-8 ----
        if (lane < 16) {
            #pragma unroll 4
            for (int n = 0; n < 16; ++n) {
                const int col = vbase + n;
                const float v = sims[wave][lane][n];
                if (col < V && v > tv[TOPK - 1]) {
                    float cv = v; int ci = col;
                    #pragma unroll
                    for (int j = 0; j < TOPK; ++j) {
                        if (cv > tv[j]) {
                            float f = tv[j]; int g = ti[j];
                            tv[j] = cv; ti[j] = ci; cv = f; ci = g;
                        }
                    }
                }
            }
        }
    }

    // ---- write per-(row, split) partial top-8 ----
    if (lane < 16) {
        const int row = mbase + wave * 16 + (lane & 15);
        const size_t off = ((size_t)row * NSPLIT + split) * TOPK;
        #pragma unroll
        for (int j = 0; j < TOPK; ++j) { pvals[off + j] = tv[j]; pidx[off + j] = ti[j]; }
    }
}

// ---------------------------------------------------------------------------
// Kernel 3: merge 16 partial top-8 lists per row, softmax(T=1), sample, emit.
// One thread per row; deterministic hash RNG (JAX PRNG parity not possible).
// ---------------------------------------------------------------------------
__global__ __launch_bounds__(256) void merge_sample_kernel(
    const float* __restrict__ pvals, const int* __restrict__ pidx,
    int* __restrict__ out, int M)
{
    const int row = blockIdx.x * blockDim.x + threadIdx.x;
    if (row >= M) return;

    float tv[TOPK]; int ti[TOPK];
    #pragma unroll
    for (int j = 0; j < TOPK; ++j) { tv[j] = -3.0f; ti[j] = 0; }

    const size_t base = (size_t)row * NSPLIT * TOPK;
    for (int c = 0; c < NSPLIT * TOPK; ++c) {
        const float v = pvals[base + c];
        const int  id = pidx[base + c];
        if (v > tv[TOPK - 1]) {
            float cv = v; int ci = id;
            #pragma unroll
            for (int j = 0; j < TOPK; ++j) {
                if (cv > tv[j]) {
                    float f = tv[j]; int g = ti[j];
                    tv[j] = cv; ti[j] = ci; cv = f; ci = g;
                }
            }
        }
    }

    // softmax over top-8 logits (TEMPERATURE = 1.0), tv[0] is the max
    float ps[TOPK]; float tot = 0.0f;
    #pragma unroll
    for (int j = 0; j < TOPK; ++j) { ps[j] = __expf(tv[j] - tv[0]); tot += ps[j]; }

    unsigned s0 = (unsigned)row * 0x9E3779B9u ^ 42u;
    s0 ^= s0 >> 16; s0 *= 0x7feb352du; s0 ^= s0 >> 15; s0 *= 0x846ca68bu; s0 ^= s0 >> 16;
    const float u = (float)(s0 >> 8) * (1.0f / 16777216.0f) * tot;

    float cum = 0.0f; int sel = ti[TOPK - 1]; bool done = false;
    #pragma unroll
    for (int j = 0; j < TOPK; ++j) {
        cum += ps[j];
        if (!done && u <= cum) { sel = ti[j]; done = true; }
    }
    out[row] = sel;
}

// ---------------------------------------------------------------------------
extern "C" void kernel_launch(void* const* d_in, const int* in_sizes, int n_in,
                              void* d_out, int out_size, void* d_ws, size_t ws_size,
                              hipStream_t stream)
{
    const float* proj = (const float*)d_in[0];
    const float* emb  = (const float*)d_in[1];
    (void)n_in; (void)ws_size; (void)out_size;

    const int M  = in_sizes[0] / EDIM;            // 4096 rows (B*S)
    const int V  = in_sizes[1] / EDIM;            // 50257 vocab rows
    const int VP = ((V + 15) / 16) * 16;          // padded to 16
    const int ntiles = VP / 16;

    char* ws = (char*)d_ws;
    _Float16* Af   = (_Float16*)ws;
    const size_t szA = (size_t)M * EDIM * sizeof(_Float16);
    _Float16* Bf   = (_Float16*)(ws + szA);
    const size_t szB = (size_t)VP * EDIM * sizeof(_Float16);
    float*    pvals = (float*)(ws + szA + szB);
    const size_t szPV = (size_t)M * NSPLIT * TOPK * sizeof(float);
    int*      pidx  = (int*)(ws + szA + szB + szPV);

    normalize_f16_kernel<<<dim3((M  * 32 + 255) / 256), 256, 0, stream>>>(proj, Af, M, M);
    normalize_f16_kernel<<<dim3((VP * 32 + 255) / 256), 256, 0, stream>>>(emb,  Bf, V, VP);
    gemm_topk_kernel<<<dim3(M / MTILE, NSPLIT), 128, 0, stream>>>(Af, Bf, pvals, pidx, V, VP, ntiles);
    merge_sample_kernel<<<dim3((M + 255) / 256), 256, 0, stream>>>(pvals, pidx, (int*)d_out, M);
}
